// Block_37443524886849
// MI455X (gfx1250) — matrix operand order repaired and checked by reading, not compiled, not run on hardware
//
#include <hip/hip_runtime.h>
#include <cmath>

// ---------------------------------------------------------------------------
// Transformer block for MI455X (gfx1250, wave32, WMMA).
// All matmuls run through v_wmma_f32_16x16x32_bf16 (bf16 A/B, f32 accum).
// GEMMs stage 128x128 block tiles in LDS with double-buffered async
// global->LDS copies (ASYNCcnt) overlapped with WMMA compute.
// ---------------------------------------------------------------------------

typedef __attribute__((ext_vector_type(16))) __bf16 v16bf;
typedef __attribute__((ext_vector_type(4)))  __bf16 v4bf;
typedef __attribute__((ext_vector_type(8)))  float  v8f;
typedef __attribute__((ext_vector_type(4)))  int    v4i;

#define BB 8
#define TT 1024
#define EMBD 1024
#define NH 16
#define HD 64
#define FF 4096
#define MM (BB * TT)   // 8192 rows

#ifndef __has_builtin
#define __has_builtin(x) 0
#endif

#if __has_builtin(__builtin_amdgcn_global_load_async_to_lds_b128)
#define HAVE_ASYNC_LDS 1
#else
#define HAVE_ASYNC_LDS 0
#endif
#if __has_builtin(__builtin_amdgcn_s_wait_asynccnt)
#define HAVE_WAIT_ASYNC 1
#else
#define HAVE_WAIT_ASYNC 0
#endif

#if HAVE_ASYNC_LDS
typedef __attribute__((address_space(1))) v4i as1_v4i;
typedef __attribute__((address_space(3))) v4i as3_v4i;
#endif

union FragBF {
    v16bf v;
    uint4 q[2];
};

__device__ __forceinline__ v8f wmma_bf16(v16bf a, v16bf b, v8f c) {
    // D = A(16x32 bf16) * B(32x16 bf16) + C(16x16 f32)
    return __builtin_amdgcn_wmma_f32_16x16x32_bf16(
        /*neg_a=*/false, a, /*neg_b=*/false, b,
        /*c_mod=*/(short)0, c, /*reuse_a=*/false, /*reuse_b=*/false);
}

// 16B global -> LDS copy: async (ASYNCcnt) when available, sync fallback.
__device__ __forceinline__ void copy16_g2l(const __bf16* __restrict__ gsrc,
                                           __bf16* __restrict__ ldst) {
#if HAVE_ASYNC_LDS
    __builtin_amdgcn_global_load_async_to_lds_b128(
        (as1_v4i*)gsrc, (as3_v4i*)ldst, 0, 0);
#else
    *(uint4*)ldst = *(const uint4*)gsrc;
#endif
}

__device__ __forceinline__ void wait_g2l() {
#if HAVE_ASYNC_LDS
#if HAVE_WAIT_ASYNC
    __builtin_amdgcn_s_wait_asynccnt(0);
#else
    asm volatile("s_wait_asynccnt 0" ::: "memory");
#endif
#endif
    // sync fallback: __syncthreads() after this call orders the ds_stores.
}

// ---------------------------------------------------------------------------
// fp32 -> bf16 conversion (4 elements/thread, b128 in / b64 out)
// ---------------------------------------------------------------------------
__global__ __launch_bounds__(256) void cvt_f32_bf16(
    const float* __restrict__ src, __bf16* __restrict__ dst, int n4) {
    int i = blockIdx.x * blockDim.x + threadIdx.x;
    if (i < n4) {
        float4 f = ((const float4*)src)[i];
        v4bf o;
        o.x = (__bf16)f.x; o.y = (__bf16)f.y; o.z = (__bf16)f.z; o.w = (__bf16)f.w;
        ((v4bf*)dst)[i] = o;
    }
}

// ---------------------------------------------------------------------------
// LayerNorm over E=1024, one block (256 thr) per row, bf16 output.
// ---------------------------------------------------------------------------
__global__ __launch_bounds__(256) void layernorm_bf16(
    const float* __restrict__ x, const float* __restrict__ g,
    const float* __restrict__ b, __bf16* __restrict__ out) {
    const int row = blockIdx.x;
    const int tid = threadIdx.x;
    const float* xr = x + (size_t)row * EMBD;
    float4 xv = ((const float4*)xr)[tid];

    float s  = xv.x + xv.y + xv.z + xv.w;
    float s2 = xv.x * xv.x + xv.y * xv.y + xv.z * xv.z + xv.w * xv.w;

    __shared__ float r1[256], r2[256];
    r1[tid] = s; r2[tid] = s2;
    __syncthreads();
    for (int st = 128; st > 0; st >>= 1) {
        if (tid < st) { r1[tid] += r1[tid + st]; r2[tid] += r2[tid + st]; }
        __syncthreads();
    }
    const float mu  = r1[0] * (1.0f / EMBD);
    const float var = r2[0] * (1.0f / EMBD) - mu * mu;
    const float inv = rsqrtf(var + 1e-5f);

    const int base = tid * 4;
    __bf16* orow = out + (size_t)row * EMBD;
    v4bf o;
    o.x = (__bf16)((xv.x - mu) * inv * g[base + 0] + b[base + 0]);
    o.y = (__bf16)((xv.y - mu) * inv * g[base + 1] + b[base + 1]);
    o.z = (__bf16)((xv.z - mu) * inv * g[base + 2] + b[base + 2]);
    o.w = (__bf16)((xv.w - mu) * inv * g[base + 3] + b[base + 3]);
    ((v4bf*)orow)[tid] = o;
}

// ---------------------------------------------------------------------------
// k [B,T,H,D] -> kt [B,H,D,T]  (so K^T B-fragments are contiguous loads)
// ---------------------------------------------------------------------------
__global__ __launch_bounds__(256) void transpose_kT(
    const __bf16* __restrict__ kin, __bf16* __restrict__ kout) {
    size_t idx = (size_t)blockIdx.x * blockDim.x + threadIdx.x; // over B*H*D*T
    int t = (int)(idx & (TT - 1));
    int d = (int)((idx >> 10) & (HD - 1));
    int h = (int)((idx >> 16) & (NH - 1));
    int b = (int)(idx >> 20);
    kout[idx] = kin[(((size_t)b * TT + t) * NH + h) * HD + d];
}

// ---------------------------------------------------------------------------
// WMMA GEMM: C[M,N] = A[M,K] (bf16 row-major) @ B[K,N] (bf16 row-major)
// Block tile 128(M) x 128(N), K-step 32, LDS double-buffered with async
// global->LDS staging. 8 waves arranged 2(M) x 4(N); wave tile 64 x 32
// -> 8 WMMAs per K-step per wave from LDS fragments.
//   A fragment (16x32): lane holds row m=lane&15; K striped over halves.
//   B fragment (32x16): lane holds column n=lane&15; K striped over halves as in A.
// ---------------------------------------------------------------------------
enum { EPI_BF16 = 0, EPI_BIAS_RES_F32 = 1, EPI_BIAS_GELU_BF16 = 2 };

#define PADA 40    // LDS row stride (elements) for A tile [128 x 32]
#define PADB 136   // LDS row stride (elements) for B tile [32 x 128]

template <int EPI>
__global__ __launch_bounds__(256) void gemm_bf16_wmma(
    const __bf16* __restrict__ A, const __bf16* __restrict__ Bw,
    const float* __restrict__ bias, const float* __restrict__ res,
    void* __restrict__ Cout, int M, int N, int K) {
    __shared__ __align__(16) __bf16 lA[2][128 * PADA];
    __shared__ __align__(16) __bf16 lB[2][32 * PADB];

    const int tid  = threadIdx.x;
    const int lane = tid & 31;
    const int wave = tid >> 5;
    const int half = lane >> 4;
    const int nl   = lane & 15;
    const int wm = (wave >> 2) * 64;   // 0 / 64
    const int wn = (wave & 3) * 32;    // 0 / 32 / 64 / 96
    const int m0 = blockIdx.y * 128;
    const int n0 = blockIdx.x * 128;

    // Each thread stages 2 x 16B of A and 2 x 16B of B per K-slice.
    auto load_tiles = [&](int ks, int buf) {
#pragma unroll
        for (int j = 0; j < 2; ++j) {
            const int c = tid * 2 + j;                 // 0..511
            const int ar = c >> 2, ac = (c & 3) * 8;   // A: 128 rows x 4 chunks
            copy16_g2l(A + (size_t)(m0 + ar) * K + ks + ac,
                       &lA[buf][ar * PADA + ac]);
            const int br = c >> 4, bc = (c & 15) * 8;  // B: 32 rows x 16 chunks
            copy16_g2l(Bw + (size_t)(ks + br) * N + n0 + bc,
                       &lB[buf][br * PADB + bc]);
        }
    };

    const v8f vzero = {0.f, 0.f, 0.f, 0.f, 0.f, 0.f, 0.f, 0.f};
    v8f acc[4][2];
#pragma unroll
    for (int r = 0; r < 4; ++r) { acc[r][0] = vzero; acc[r][1] = vzero; }

    load_tiles(0, 0);

    const int nk = K / 32;
    for (int t = 0; t < nk; ++t) {
        wait_g2l();
        __syncthreads();                 // staged slice visible to all waves
        if (t + 1 < nk) load_tiles((t + 1) * 32, (t + 1) & 1);  // prefetch

        const __bf16* la = lA[t & 1];
        const __bf16* lb = lB[t & 1];

        FragBF bfr[2];
#pragma unroll
        for (int q = 0; q < 2; ++q) {
            const __bf16* bp = lb + wn + q * 16 + nl;
#pragma unroll
            for (int s = 0; s < 16; ++s) bfr[q].v[s] = bp[(half * 8 + s + (s & 8)) * PADB];
        }
#pragma unroll
        for (int r = 0; r < 4; ++r) {
            FragBF af;
            const __bf16* ap = la + (wm + r * 16 + nl) * PADA + half * 8;
            af.q[0] = *(const uint4*)(ap);
            af.q[1] = *(const uint4*)(ap + 16);
            acc[r][0] = wmma_bf16(af.v, bfr[0].v, acc[r][0]);
            acc[r][1] = wmma_bf16(af.v, bfr[1].v, acc[r][1]);
        }
        __syncthreads();                 // all reads done before buf reuse
    }

#pragma unroll
    for (int r = 0; r < 4; ++r) {
#pragma unroll
        for (int q = 0; q < 2; ++q) {
#pragma unroll
            for (int i = 0; i < 8; ++i) {
                const int row = m0 + wm + r * 16 + i + 8 * half;
                const int col = n0 + wn + q * 16 + nl;
                const size_t idx = (size_t)row * N + col;
                const float val = acc[r][q][i];
                if (EPI == EPI_BF16) {
                    ((__bf16*)Cout)[idx] = (__bf16)val;
                } else if (EPI == EPI_BIAS_RES_F32) {
                    ((float*)Cout)[idx] = val + bias[col] + res[idx];
                } else { // bias + exact GELU -> bf16
                    const float tv = val + bias[col];
                    const float gl = 0.5f * tv * (1.0f + erff(tv * 0.70710678118654752f));
                    ((__bf16*)Cout)[idx] = (__bf16)gl;
                }
            }
        }
    }
}

// ---------------------------------------------------------------------------
// Flash-style causal attention. Wave = one 16-query tile; loop over 32-key
// chunks. S = Q*K^T via WMMA, online softmax, P routed through LDS to swap
// C-layout -> A-layout, then O += P*V via WMMA. Output bf16 [B,T,E].
// ---------------------------------------------------------------------------
__global__ __launch_bounds__(128) void attention_wmma(
    const __bf16* __restrict__ qm, const __bf16* __restrict__ kt,
    const __bf16* __restrict__ vm, __bf16* __restrict__ attn) {
    const int lane = threadIdx.x & 31;
    const int wave = threadIdx.x >> 5;
    const int half = lane >> 4;
    const int nl   = lane & 15;
    const int bh = blockIdx.y;
    const int b = bh >> 4;
    const int h = bh & 15;
    const int q0 = (blockIdx.x * 4 + wave) * 16;

    // q/v: [B,T,H,D]; kt: [B,H,D,T]
    const __bf16* qbase = qm + ((size_t)b * TT * NH + h) * HD;
    const __bf16* vbase = vm + ((size_t)b * TT * NH + h) * HD;
    const __bf16* ktb   = kt + ((size_t)(b * NH + h)) * HD * TT;

    // Preload Q A-fragments for both 32-wide d steps.
    FragBF qa[2];
#pragma unroll
    for (int s = 0; s < 2; ++s) {
        const __bf16* ap = qbase + (size_t)(q0 + nl) * (NH * HD) + s * 32 + half * 8;
        qa[s].q[0] = *(const uint4*)(ap);
        qa[s].q[1] = *(const uint4*)(ap + 16);
    }

    const v8f vzero = {0.f, 0.f, 0.f, 0.f, 0.f, 0.f, 0.f, 0.f};
    v8f o[4];
    o[0] = vzero; o[1] = vzero; o[2] = vzero; o[3] = vzero;
    float mrow[8], lrow[8];
#pragma unroll
    for (int i = 0; i < 8; ++i) { mrow[i] = -1e30f; lrow[i] = 0.f; }

    __shared__ __align__(16) __bf16 plds[4][16 * 40]; // stride 40: 16B align + pad
    __bf16* pl = plds[wave];

    const int nch = (q0 + 15) / 32 + 1;
    for (int c = 0; c < nch; ++c) {
        const int k0 = c * 32;
        v8f s0 = vzero, s1 = vzero;
#pragma unroll
        for (int ds = 0; ds < 2; ++ds) {
            const __bf16* bp = ktb + (size_t)(ds * 32) * TT + k0 + nl;
            FragBF kf0, kf1;
#pragma unroll
            for (int s = 0; s < 16; ++s) {
                const size_t kk = (size_t)(half * 8 + s + (s & 8)) * TT;
                kf0.v[s] = bp[kk];
                kf1.v[s] = bp[kk + 16];
            }
            s0 = wmma_bf16(qa[ds].v, kf0.v, s0);
            s1 = wmma_bf16(qa[ds].v, kf1.v, s1);
        }

        // scale + causal mask + online softmax (rows live in lane halves)
        const float scale = 0.125f; // 1/sqrt(64)
#pragma unroll
        for (int i = 0; i < 8; ++i) {
            const int qrow = q0 + i + 8 * half;
            float v0 = (k0 + nl      <= qrow) ? s0[i] * scale : -1e30f;
            float v1 = (k0 + 16 + nl <= qrow) ? s1[i] * scale : -1e30f;
            float mx = fmaxf(v0, v1);
#pragma unroll
            for (int off = 8; off > 0; off >>= 1) mx = fmaxf(mx, __shfl_xor(mx, off, 16));
            const float mnew  = fmaxf(mrow[i], mx);
            const float alpha = __expf(mrow[i] - mnew);
            const float p0 = __expf(v0 - mnew);
            const float p1 = __expf(v1 - mnew);
            float rs = p0 + p1;
#pragma unroll
            for (int off = 8; off > 0; off >>= 1) rs += __shfl_xor(rs, off, 16);
            lrow[i] = lrow[i] * alpha + rs;
            mrow[i] = mnew;
            o[0][i] *= alpha; o[1][i] *= alpha; o[2][i] *= alpha; o[3][i] *= alpha;
            // stash P (bf16) into LDS in [row][key] order
            pl[(i + 8 * half) * 40 + nl]      = (__bf16)p0;
            pl[(i + 8 * half) * 40 + 16 + nl] = (__bf16)p1;
        }
        asm volatile("s_wait_dscnt 0" ::: "memory");

        // reload P as A-fragment (16 x 32)
        FragBF pf;
        {
            const __bf16* ap = pl + nl * 40 + half * 8;
            pf.q[0] = *(const uint4*)(ap);
            pf.q[1] = *(const uint4*)(ap + 16);
        }
        // O += P @ V : V B-fragment lane holds column d = nl, 16 keys (table k order)
#pragma unroll
        for (int dt = 0; dt < 4; ++dt) {
            const __bf16* vp = vbase + (size_t)k0 * (NH * HD) + dt * 16 + nl;
            FragBF vf;
#pragma unroll
            for (int s = 0; s < 16; ++s) vf.v[s] = vp[(size_t)(half * 8 + s + (s & 8)) * (NH * HD)];
            o[dt] = wmma_bf16(pf.v, vf.v, o[dt]);
        }
    }

    // normalize and store bf16 [B,T,E]
#pragma unroll
    for (int dt = 0; dt < 4; ++dt) {
#pragma unroll
        for (int i = 0; i < 8; ++i) {
            const int t = q0 + i + 8 * half;
            attn[((size_t)(b * TT + t)) * EMBD + h * HD + dt * 16 + nl] =
                (__bf16)(o[dt][i] / lrow[i]);
        }
    }
}

// ---------------------------------------------------------------------------
// Host-side orchestration
// ---------------------------------------------------------------------------
extern "C" void kernel_launch(void* const* d_in, const int* in_sizes, int n_in,
                              void* d_out, int out_size, void* d_ws, size_t ws_size,
                              hipStream_t stream) {
    (void)in_sizes; (void)n_in; (void)out_size; (void)ws_size;

    const float* x      = (const float*)d_in[0];
    const float* ln1_g  = (const float*)d_in[1];
    const float* ln1_b  = (const float*)d_in[2];
    const float* wq     = (const float*)d_in[3];
    const float* wk     = (const float*)d_in[4];
    const float* wv     = (const float*)d_in[5];
    const float* w_proj = (const float*)d_in[6];
    const float* b_proj = (const float*)d_in[7];
    const float* ln2_g  = (const float*)d_in[8];
    const float* ln2_b  = (const float*)d_in[9];
    const float* w1     = (const float*)d_in[10];
    const float* b1     = (const float*)d_in[11];
    const float* w2     = (const float*)d_in[12];
    const float* b2     = (const float*)d_in[13];
    float* out_final    = (float*)d_out;

    char* ws = (char*)d_ws;
    size_t off = 0;
    auto take = [&](size_t bytes) {
        char* p = ws + off;
        off += (bytes + 255) & ~(size_t)255;
        return (void*)p;
    };

    __bf16* wq_b = (__bf16*)take((size_t)EMBD * EMBD * 2);
    __bf16* wk_b = (__bf16*)take((size_t)EMBD * EMBD * 2);
    __bf16* wv_b = (__bf16*)take((size_t)EMBD * EMBD * 2);
    __bf16* wp_b = (__bf16*)take((size_t)EMBD * EMBD * 2);
    __bf16* w1_b = (__bf16*)take((size_t)EMBD * FF * 2);
    __bf16* w2_b = (__bf16*)take((size_t)FF * EMBD * 2);
    __bf16* h_b  = (__bf16*)take((size_t)MM * EMBD * 2);
    __bf16* q_b  = (__bf16*)take((size_t)MM * EMBD * 2);
    __bf16* k_b  = (__bf16*)take((size_t)MM * EMBD * 2);
    __bf16* v_b  = (__bf16*)take((size_t)MM * EMBD * 2);
    __bf16* kt_b = (__bf16*)take((size_t)MM * EMBD * 2);
    __bf16* at_b = (__bf16*)take((size_t)MM * EMBD * 2);
    float*  out1 = (float*) take((size_t)MM * EMBD * 4);
    __bf16* h2_b = (__bf16*)take((size_t)MM * EMBD * 2);
    __bf16* g_b  = (__bf16*)take((size_t)MM * FF * 2);

    // 1) weights -> bf16
    auto cvt = [&](const float* s, __bf16* d, size_t n) {
        int n4 = (int)(n / 4);
        cvt_f32_bf16<<<(n4 + 255) / 256, 256, 0, stream>>>(s, d, n4);
    };
    cvt(wq, wq_b, (size_t)EMBD * EMBD);
    cvt(wk, wk_b, (size_t)EMBD * EMBD);
    cvt(wv, wv_b, (size_t)EMBD * EMBD);
    cvt(w_proj, wp_b, (size_t)EMBD * EMBD);
    cvt(w1, w1_b, (size_t)EMBD * FF);
    cvt(w2, w2_b, (size_t)FF * EMBD);

    // 2) LN1
    layernorm_bf16<<<MM, 256, 0, stream>>>(x, ln1_g, ln1_b, h_b);

    // 3) QKV projections (WMMA)
    dim3 gQ(EMBD / 128, MM / 128);
    gemm_bf16_wmma<EPI_BF16><<<gQ, 256, 0, stream>>>(h_b, wq_b, nullptr, nullptr, q_b, MM, EMBD, EMBD);
    gemm_bf16_wmma<EPI_BF16><<<gQ, 256, 0, stream>>>(h_b, wk_b, nullptr, nullptr, k_b, MM, EMBD, EMBD);
    gemm_bf16_wmma<EPI_BF16><<<gQ, 256, 0, stream>>>(h_b, wv_b, nullptr, nullptr, v_b, MM, EMBD, EMBD);

    // 4) K -> K^T per head
    transpose_kT<<<(BB * NH * HD * TT) / 256, 256, 0, stream>>>(k_b, kt_b);

    // 5) causal flash attention (WMMA)
    attention_wmma<<<dim3(TT / 64, BB * NH), 128, 0, stream>>>(q_b, kt_b, v_b, at_b);

    // 6) out = x + attn @ w_proj + b_proj
    gemm_bf16_wmma<EPI_BIAS_RES_F32><<<gQ, 256, 0, stream>>>(at_b, wp_b, b_proj, x, out1, MM, EMBD, EMBD);

    // 7) LN2
    layernorm_bf16<<<MM, 256, 0, stream>>>(out1, ln2_g, ln2_b, h2_b);

    // 8) gelu(h2 @ w1 + b1) -> bf16
    dim3 gU(FF / 128, MM / 128);
    gemm_bf16_wmma<EPI_BIAS_GELU_BF16><<<gU, 256, 0, stream>>>(h2_b, w1_b, b1, nullptr, g_b, MM, FF, EMBD);

    // 9) final = out + g @ w2 + b2
    gemm_bf16_wmma<EPI_BIAS_RES_F32><<<gQ, 256, 0, stream>>>(g_b, w2_b, b2, out1, out_final, MM, EMBD, FF);
}
